// ConvAttention2d_38079180046956
// MI455X (gfx1250) — matrix operand order repaired and checked by reading, not compiled, not run on hardware
//
#include <hip/hip_runtime.h>
#include <math.h>

// ---------------------------------------------------------------------------
// ConvAttention2d fused pipeline for MI455X (gfx1250, wave32, WMMA)
//
// Stage 1: per-patch depthwise 3x3 conv  -> sa (bf16, NHWC) in workspace
// Stage 2: dense 96->96 3x3 conv as implicit GEMM on v_wmma_f32_16x16x32_bf16
//          (M = B*H*W pixels, N = 96, K = 9*96 = 864 = 27 chunks of 32),
//          exact-erf GELU, fp32 NCHW out.
//
// Roofline: HBM ~0.92 GB total (~40us @ 23.3 TB/s); sa (154 MB bf16) fits the
// 192 MB L2 so the 9x tap reuse is L2-served; 133 GFLOP on bf16 WMMA lands in
// the same tens-of-us regime -> balanced.
//
// Workspace layout: [0, 166KB)   fragment-major packed bf16 weights (L2-resident)
//                   [1MB, +154MB) sa bf16 NHWC
// ---------------------------------------------------------------------------

#define B_    16
#define C_    96
#define H_    224
#define W_    224
#define COUT_ 96
#define GW_   14
#define P_    196          // 14*14 patches
#define NKCH  27           // 864 / 32 K-chunks

typedef __bf16 bf16_t;
typedef bf16_t        v16bf __attribute__((ext_vector_type(16)));
typedef float         v8f   __attribute__((ext_vector_type(8)));
typedef unsigned int  u32x4 __attribute__((ext_vector_type(4)));

__device__ __forceinline__ unsigned int f2bf(float f) {
  unsigned int u = __float_as_uint(f);
  u = u + 0x7FFFu + ((u >> 16) & 1u);       // round-to-nearest-even
  return u >> 16;
}

// ---------------------------------------------------------------------------
// Kernel 0: pack output_filters (COUT, C, 3, 3) fp32 into FRAGMENT-MAJOR bf16
// B-matrix storage.  GEMM K index: k = tap*96 + c (tap = kh*3+kw).
// Fragment entry fr = ((j*6 + nt)*2 + g)*16 + m holds 8 dwords; dword v packs
// bf16 K-rows (2*kp, 2*kp+1) with kp = j*16 + (v>>2)*8 + g*4 + (v&3), column n = nt*16 + m.
// => in kernel 2, each lane's B fragment is two contiguous b128 loads.
// Total 41472 dwords = 162 KB.
// ---------------------------------------------------------------------------
__global__ __launch_bounds__(256) void pack_weights(const float* __restrict__ of,
                                                    unsigned int* __restrict__ wfrag) {
  int tid = blockIdx.x * 256 + threadIdx.x;
  if (tid >= 432 * 96) return;
  int v  = tid & 7;
  int fr = tid >> 3;
  int m  = fr & 15;
  int g  = (fr >> 4) & 1;
  int nt = (fr >> 5) % 6;
  int j  = fr / 192;
  int kp  = j * 16 + (v >> 2) * 8 + g * 4 + (v & 3);
  int n   = nt * 16 + m;
  int k0  = kp * 2;
  int tap = k0 / 96;
  int c   = k0 % 96;                          // even, so c+1 shares the tap
  float f0 = of[((size_t)n * 96 + c)     * 9 + tap];
  float f1 = of[((size_t)n * 96 + c + 1) * 9 + tap];
  wfrag[tid] = f2bf(f0) | (f2bf(f1) << 16);
}

// ---------------------------------------------------------------------------
// Kernel 1: per-patch depthwise conv.  One block = one (batch, patch).
// 256 threads = the 16x16 pixels of the patch.  x tile staged in LDS
// (4 channels/iter), result packed bf16 into LDS, final NHWC store is
// 192 contiguous bytes per thread (fully coalesced across the row).
// ---------------------------------------------------------------------------
__global__ __launch_bounds__(256) void self_conv(const float* __restrict__ x,
                                                 const float* __restrict__ pf,
                                                 unsigned int* __restrict__ sa32) {
  __shared__ float        xs[4][256];
  __shared__ unsigned int ys[256 * 48];        // 256 px * 96 ch (bf16 pairs) = 48KB

  int t     = threadIdx.x;
  int blk   = blockIdx.x;
  int b     = blk / P_;
  int patch = blk % P_;
  int gh = patch / GW_, gw = patch % GW_;
  int pr = t >> 4,      pc = t & 15;
  int hh = gh * 16 + pr, ww = gw * 16 + pc;

  const float* xbase = x + (((size_t)b * C_) * H_ + hh) * W_ + ww;

  for (int cb = 0; cb < 24; ++cb) {
    int c0 = cb * 4;
#pragma unroll
    for (int cc = 0; cc < 4; ++cc)
      xs[cc][t] = xbase[(size_t)(c0 + cc) * (H_ * W_)];
    __syncthreads();

    float r[4];
#pragma unroll
    for (int cc = 0; cc < 4; ++cc) {
      int c = c0 + cc;
      const float* f = pf + ((size_t)c * P_ + patch) * 9;
      float a = 0.f;
#pragma unroll
      for (int kh = 0; kh < 3; ++kh) {
        int ih = pr + kh - 1;
#pragma unroll
        for (int kw = 0; kw < 3; ++kw) {
          int iw = pc + kw - 1;
          if (ih >= 0 && ih < 16 && iw >= 0 && iw < 16)
            a += xs[cc][ih * 16 + iw] * f[kh * 3 + kw];
        }
      }
      r[cc] = a;
    }
    ys[t * 48 + cb * 2 + 0] = f2bf(r[0]) | (f2bf(r[1]) << 16);
    ys[t * 48 + cb * 2 + 1] = f2bf(r[2]) | (f2bf(r[3]) << 16);
    __syncthreads();
  }

  // NHWC bf16 store: 192B contiguous per thread, contiguous across the row.
  size_t pix = ((size_t)b * H_ + hh) * W_ + ww;
  u32x4* dst = (u32x4*)(sa32 + pix * 48);
  const u32x4* src = (const u32x4*)&ys[t * 48];
#pragma unroll
  for (int i = 0; i < 12; ++i) dst[i] = src[i];
}

// ---------------------------------------------------------------------------
// Kernel 2: implicit-GEMM conv + exact GELU.
// Block = 256 threads = 8 waves; wave owns 16 consecutive pixels (one row
// segment, since 224 % 16 == 0) x all 96 couts -> 6 f32 accumulators.
// Fully-unrolled K loop; INTERIOR template removes all bounds predication on
// the ~85% of tiles with no image-border taps.
// ---------------------------------------------------------------------------
template <bool INTERIOR>
__device__ __forceinline__ void gemm_k(const unsigned short* __restrict__ sa,
                                       const u32x4* __restrict__ wq,
                                       v8f (&acc)[6],
                                       int b, int h, int w0, int m, int g) {
#pragma unroll
  for (int j = 0; j < NKCH; ++j) {
    const int tap = j / 3;                    // constants after full unroll
    const int c0  = (j % 3) * 32;
    const int dh  = tap / 3 - 1, dw = tap % 3 - 1;
    int hh = h + dh;
    int ww = w0 + m + dw;

    // A fragment: 16x32 bf16 per documented lane layout; NHWC makes the 32
    // K-channels contiguous => two b128 loads per lane.
    union { u32x4 q[2]; v16bf v; } au;
    if (INTERIOR) {
      const char* cp = (const char*)(sa + (((size_t)b * H_ + hh) * W_ + ww) * C_ + c0);
      au.q[0] = *(const u32x4*)(cp + g * 16);        // VGPRs 0..3: K = 8g..8g+7
      au.q[1] = *(const u32x4*)(cp + 32 + g * 16);   // VGPRs 4..7: K = 16+8g..
    } else {
      au.q[0] = (u32x4)0u;
      au.q[1] = (u32x4)0u;
      if (hh >= 0 && hh < H_ && ww >= 0 && ww < W_) {
        const char* cp = (const char*)(sa + (((size_t)b * H_ + hh) * W_ + ww) * C_ + c0);
        au.q[0] = *(const u32x4*)(cp + g * 16);
        au.q[1] = *(const u32x4*)(cp + 32 + g * 16);
      }
    }

#pragma unroll
    for (int nt = 0; nt < 6; ++nt) {
      // Fragment-major weights: 2 contiguous b128 loads per lane (L2-resident).
      const u32x4* wf = wq + (size_t)(((j * 6 + nt) * 2 + g) * 16 + m) * 2;
      union { u32x4 q[2]; v16bf v; } bu;
      bu.q[0] = wf[0];
      bu.q[1] = wf[1];
      acc[nt] = __builtin_amdgcn_wmma_f32_16x16x32_bf16(
          false, au.v, false, bu.v, (short)0, acc[nt], false, false);
    }
  }
}

__global__ __launch_bounds__(256) void conv_gemm_gelu(
    const unsigned short* __restrict__ sa, const u32x4* __restrict__ wq,
    float* __restrict__ out) {
  __shared__ float sbuf[8][16][17];            // [wave][N][M], padded: no conflicts

  int lane  = threadIdx.x & 31;
  int wave  = threadIdx.x >> 5;
  int mtile = blockIdx.x * 8 + wave;           // 50176 tiles total
  int p0    = mtile * 16;
  int b     = p0 / (H_ * W_);
  int rem   = p0 % (H_ * W_);
  int h     = rem / W_;
  int w0    = rem % W_;                        // multiple of 16
  int m     = lane & 15;                       // A row / B,C,D column
  int g     = lane >> 4;                       // K half-select per ISA layout

  v8f acc[6];
#pragma unroll
  for (int i = 0; i < 6; ++i)
#pragma unroll
    for (int r = 0; r < 8; ++r) acc[i][r] = 0.0f;

  // Wave-uniform interior test: every (dh,dw) tap of every lane stays in-image.
  bool interior = (h >= 1) && (h <= H_ - 2) && (w0 >= 16) && (w0 <= W_ - 32);
  if (interior)
    gemm_k<true>(sa, wq, acc, b, h, w0, m, g);
  else
    gemm_k<false>(sa, wq, acc, b, h, w0, m, g);

  // Exact-erf GELU + LDS transpose so NCHW stores are 64B contiguous.
  const float kInvSqrt2 = 0.70710678118654752f;
#pragma unroll
  for (int nt = 0; nt < 6; ++nt) {
    __syncthreads();
#pragma unroll
    for (int r = 0; r < 8; ++r) {              // C layout: N=m, M=r+8g
      float xv = acc[nt][r];
      sbuf[wave][m][r + 8 * g] = 0.5f * xv * (1.0f + erff(xv * kInvSqrt2));
    }
    __syncthreads();
#pragma unroll
    for (int r2 = 0; r2 < 8; ++r2) {
      int nl = r2 * 2 + g;
      out[(((size_t)b * COUT_ + nt * 16 + nl) * H_ + h) * W_ + w0 + m] =
          sbuf[wave][nl][m];
    }
  }
}

// ---------------------------------------------------------------------------
extern "C" void kernel_launch(void* const* d_in, const int* in_sizes, int n_in,
                              void* d_out, int out_size, void* d_ws, size_t ws_size,
                              hipStream_t stream) {
  const float* x  = (const float*)d_in[0];   // (16,96,224,224)
  const float* pf = (const float*)d_in[1];   // (96,196,1,3,3)
  const float* of = (const float*)d_in[2];   // (96,96,3,3)
  float* out = (float*)d_out;                // (16,96,224,224) fp32

  unsigned int*   wfrag = (unsigned int*)d_ws;                       // 162 KB
  unsigned int*   sa32  = (unsigned int*)((char*)d_ws + (1 << 20));  // 154 MB bf16
  unsigned short* sa16  = (unsigned short*)sa32;

  pack_weights<<<(432 * 96 + 255) / 256, 256, 0, stream>>>(of, wfrag);
  self_conv<<<B_ * P_, 256, 0, stream>>>(x, pf, sa32);
  conv_gemm_gelu<<<(B_ * H_ * W_ / 16) / 8, 256, 0, stream>>>(
      sa16, (const u32x4*)wfrag, out);
}
